// OSSM_23648089932280
// MI455X (gfx1250) — matrix operand order repaired and checked by reading, not compiled, not run on hardware
//
#include <hip/hip_runtime.h>
#include <hip/hip_bf16.h>

#define BS   8
#define HH   48
#define WWI  48
#define LL   2304          // HH*WWI
#define BLR  18432         // BS*LL
#define DM   96
#define DI   192
#define NS   16
#define RK   6
#define KG   4
#define CDBL 38            // RK + 2*NS
#define TCH  48            // scan staging chunk (timesteps)

typedef __attribute__((ext_vector_type(16))) _Float16 v16h;
typedef __attribute__((ext_vector_type(8)))  _Float16 v8h;
typedef __attribute__((ext_vector_type(8)))  float    v8f;
typedef unsigned int v4u __attribute__((ext_vector_type(4)));
typedef int          v8i __attribute__((ext_vector_type(8)));
typedef int          v4i __attribute__((ext_vector_type(4)));

#if __has_builtin(__builtin_amdgcn_tensor_load_to_lds) && __has_builtin(__builtin_amdgcn_s_wait_tensorcnt)
#define USE_TDM 1
#endif

__device__ inline float siluf(float x)     { return x / (1.0f + __expf(-x)); }
__device__ inline float softplusf(float x) { return (x > 20.0f) ? x : log1pf(__expf(x)); }

// ---------------- pack / convert ----------------
__global__ void k_f32_to_f16(const float* __restrict__ src, _Float16* __restrict__ dst, int n) {
  int i = blockIdx.x * blockDim.x + threadIdx.x;
  if (i < n) dst[i] = (_Float16)src[i];
}

__global__ void k_zero(float* __restrict__ p, int n) {
  int i = blockIdx.x * blockDim.x + threadIdx.x;
  if (i < n) p[i] = 0.0f;
}

// ---------------- WMMA multi-tile GEMM core (f16 in, f32 acc) ----------------
// A : row-major (M x K), lda (halves); rows clamped to rowMax for padded M tiles.
// Bt: N-major (N x K), ldb (halves)  -> lane's B tile is two 16B runs, 32B apart.
// ISA layouts (CDNA5 ISA):
//   A 16x32 f16 : lane m=lane&15 holds K in [kh,kh+8) u [16+kh,16+kh+8), kh=(lane>>4)*8
//   B 32x16 f16 : lane n=lane&15 holds K in [kh,kh+8) u [16+kh,16+kh+8), kh=(lane>>4)*8
//   C/D 16x16 f32: VGPR v -> M = v + (lane>>4)*8, N = lane&15
template <int NT>
__device__ inline void wmma_rows(const _Float16* __restrict__ A, int lda, int rowMax,
                                 const _Float16* __restrict__ Bt, int ldb,
                                 int m0, int n0, int K, int lane, v8f* acc) {
  int m = m0 + (lane & 15); if (m > rowMax) m = rowMax;
  int kh  = (lane >> 4) << 3;
  int nn  = n0 + (lane & 15);
  int kb2 = (lane >> 4) << 3;
  for (int kk = 0; kk < K; kk += 32) {
    const _Float16* Ar = A + (size_t)m * lda + kk + kh;
    v8h a0 = *(const v8h*)(Ar);
    v8h a1 = *(const v8h*)(Ar + 16);
    v16h a;
#pragma unroll
    for (int j = 0; j < 8; ++j) { a[j] = a0[j]; a[8 + j] = a1[j]; }
#pragma unroll
    for (int t = 0; t < NT; ++t) {
      const _Float16* Br = Bt + (size_t)(nn + 16 * t) * ldb + kk + kb2;
      v8h b0 = *(const v8h*)(Br);
      v8h b1 = *(const v8h*)(Br + 16);
      v16h bv;
#pragma unroll
      for (int j = 0; j < 8; ++j) { bv[j] = b0[j]; bv[8 + j] = b1[j]; }
      acc[t] = __builtin_amdgcn_wmma_f32_16x16x32_f16(false, a, false, bv, (short)0, acc[t], false, false);
    }
  }
}

// ---------------- 1) in_proj: xz = x @ W^T ; split -> xc_pre (b,d,l), z=silu (b,l,d) ----------------
__global__ __launch_bounds__(32) void k_gemm_inproj(const _Float16* __restrict__ xh,
                                                    const _Float16* __restrict__ winh,   // (384 x 96) N-major
                                                    float* __restrict__ xc_pre,
                                                    float* __restrict__ zbuf) {
  int lane = threadIdx.x;
  int m0 = blockIdx.x * 16, n0 = blockIdx.y * 64;
  v8f acc[4] = {};
  wmma_rows<4>(xh, DM, BLR - 1, winh, DM, m0, n0, DM, lane, acc);
  int ml = (lane >> 4) << 3, nl = lane & 15;
#pragma unroll
  for (int t = 0; t < 4; ++t) {
    int ng = n0 + 16 * t + nl;
#pragma unroll
    for (int v = 0; v < 8; ++v) {
      int mg = m0 + ml + v;
      int b = mg / LL, l = mg % LL;
      float val = acc[t][v];
      if (ng < DI) xc_pre[((size_t)b * DI + ng) * LL + l] = val;          // (b,d,h,w)
      else         zbuf[(size_t)mg * DI + (ng - DI)]      = siluf(val);   // (b,l,d)
    }
  }
}

// ---------------- 2) depthwise 3x3 conv (cross-correlation, SAME) + bias + SiLU ----------------
__global__ void k_conv_dw(const float* __restrict__ xin, const float* __restrict__ cw,
                          const float* __restrict__ cb, float* __restrict__ xout) {
  int i = blockIdx.x * blockDim.x + threadIdx.x;
  if (i >= BS * DI * LL) return;
  int wq = i % WWI; int t = i / WWI;
  int hq = t % HH;  t /= HH;
  int d  = t % DI;  int b = t / DI;
  const float* inp = xin + ((size_t)b * DI + d) * LL;
  const float* w   = cw + d * 9;
  float s = cb[d];
#pragma unroll
  for (int r = 0; r < 3; ++r) {
    int hy = hq + r - 1;
    if (hy < 0 || hy >= HH) continue;
#pragma unroll
    for (int c = 0; c < 3; ++c) {
      int wx = wq + c - 1;
      if (wx < 0 || wx >= WWI) continue;
      s += inp[hy * WWI + wx] * w[r * 3 + c];
    }
  }
  xout[i] = siluf(s);
}

// ---------------- 3) CrossScan: (b,d,h,w) -> xs_f16 (b,k,l,d)  [l-major for GEMM B / coalesced scan] --
__global__ void k_cross_scan(const float* __restrict__ xc, _Float16* __restrict__ xs) {
  int i = blockIdx.x * blockDim.x + threadIdx.x;
  if (i >= BS * DI * LL) return;
  int wq = i % WWI; int t = i / WWI;
  int hq = t % HH;  t /= HH;
  int d  = t % DI;  int b = t / DI;
  _Float16 hv = (_Float16)xc[i];
#pragma unroll
  for (int k = 0; k < KG; ++k) {
    int wsz = (k == 0) ? 1 : (1 << k);
    int gh = hq / wsz, ih = hq - gh * wsz;
    int gw = wq / wsz, iw = wq - gw * wsz;
    int l = ((gh * (WWI / wsz) + gw) * wsz + ih) * wsz + iw;
    xs[(((size_t)(b * KG + k)) * LL + l) * DI + d] = hv;
  }
}

// ---------------- 4) x_proj per (b,k): (38x192) @ (192xL) -> x_dbl (b,k,c,l) ----------------
__global__ __launch_bounds__(32) void k_gemm_xproj(const _Float16* __restrict__ xpw,
                                                   const _Float16* __restrict__ xs,
                                                   float* __restrict__ xdbl) {
  int lane = threadIdx.x;
  int bk = blockIdx.z;            // b*KG + k
  int k  = bk & 3;
  int n0 = blockIdx.x * 64;       // l tile group
  int m0 = blockIdx.y * 16;       // c tile (0..2 covering 48 padded rows)
  const _Float16* A  = xpw + (size_t)k * CDBL * DI;
  const _Float16* Bt = xs + (size_t)bk * LL * DI;   // (L x 192) N-major
  v8f acc[4] = {};
  wmma_rows<4>(A, DI, CDBL - 1, Bt, DI, m0, n0, DI, lane, acc);
  int ml = (lane >> 4) << 3, nl = lane & 15;
  float* out = xdbl + (size_t)bk * CDBL * LL;
#pragma unroll
  for (int t = 0; t < 4; ++t) {
    int ng = n0 + 16 * t + nl;
#pragma unroll
    for (int v = 0; v < 8; ++v) {
      int c = m0 + ml + v;
      if (c < CDBL) out[(size_t)c * LL + ng] = acc[t][v];
    }
  }
}

// ---------------- 5) selective scan (fused dt-proj + softplus + D*x + CrossMerge atomics) ----------
// grid: BS*KG*6 single-wave blocks; each wave owns 32 channels of one (b,k) stream.
// dt/B/C rows staged per 48-step chunk into LDS via the Tensor Data Mover (fallback: manual).
__global__ __launch_bounds__(32) void k_scan(const float* __restrict__ xdbl,
                                             const _Float16* __restrict__ xs,
                                             const float* __restrict__ dtw,
                                             const float* __restrict__ dtb,
                                             const float* __restrict__ Alogs,
                                             const float* __restrict__ Dsv,
                                             float* __restrict__ ybuf) {
  int lane = threadIdx.x;
  int blk  = blockIdx.x;
  int dblk = blk % 6, bk = blk / 6;
  int b = bk >> 2, k = bk & 3;
  int d  = dblk * 32 + lane;
  int kd = k * DI + d;
  __shared__ float lds[CDBL * TCH];
  const float*     xd = xdbl + (size_t)bk * CDBL * LL;
  const _Float16*  xr = xs + (size_t)bk * LL * DI + d;   // (l,d) layout: step stride DI, lane-coalesced
  float Areg[NS];
#pragma unroll
  for (int n = 0; n < NS; ++n) Areg[n] = -__expf(Alogs[kd * NS + n]);
  float wreg[RK];
#pragma unroll
  for (int r = 0; r < RK; ++r) wreg[r] = dtw[kd * RK + r];
  float bias = dtb[kd];
  float Dd   = Dsv[kd];
  float h[NS];
#pragma unroll
  for (int n = 0; n < NS; ++n) h[n] = 0.0f;
  float* yrow = ybuf + (size_t)b * LL * DI + d;          // y layout (b,l,d)
  int wsz  = (k == 0) ? 1 : (1 << k);
  int gdiv = WWI / wsz;
#ifdef USE_TDM
  unsigned ldsoff = (unsigned)(size_t)(&lds[0]);         // low 32 bits of LDS generic addr = LDS byte offset
#endif
  for (int t0 = 0; t0 < LL; t0 += TCH) {
#ifdef USE_TDM
    // D# for a 2D tile: 48 cols x 38 rows of f32, row stride LL elements.
    unsigned long long ga = (unsigned long long)(size_t)(xd + t0);
    v4u g0 = { 1u,                                    // count=1, user mode
               ldsoff,                                 // lds_addr (bytes)
               (unsigned)ga,                           // global_addr[31:0]
               (unsigned)((ga >> 32) & 0x01FFFFFFu) | 0x80000000u };  // addr[56:32] | type=2
    v8i g1 = { 0x00020000,                             // data_size=4B, no multicast/pad/iterate
               48 << 16,                               // tensor_dim0 = 48
               38 << 16,                               // tensor_dim1 = 38
               48 << 16,                               // tile_dim0 = 48
               38,                                     // tile_dim1 = 38, tile_dim2 = 0
               (int)LL,                                // tensor_dim0_stride = 2304
               0, 0 };
    v4i zz = { 0, 0, 0, 0 };
    asm volatile("s_wait_dscnt 0x0" ::: "memory");     // prior chunk's LDS reads done before DMA overwrite
#if __clang_major__ >= 23
    v8i zz8 = { 0, 0, 0, 0, 0, 0, 0, 0 };
    __builtin_amdgcn_tensor_load_to_lds(g0, g1, zz, zz, zz8, 0);
#else
    __builtin_amdgcn_tensor_load_to_lds(g0, g1, zz, zz, 0);
#endif
    __builtin_amdgcn_s_wait_tensorcnt(0);
    asm volatile("" ::: "memory");                     // LDS was written behind the compiler's back
#else
    __syncthreads();
    for (int i = lane; i < CDBL * TCH; i += 32)
      lds[i] = xd[(size_t)(i / TCH) * LL + t0 + (i % TCH)];
    __syncthreads();
#endif
    for (int tt = 0; tt < TCH; ++tt) {
      int t = t0 + tt;
      float draw = bias;
#pragma unroll
      for (int r = 0; r < RK; ++r) draw += lds[r * TCH + tt] * wreg[r];
      float delta = softplusf(draw);
      float xval  = (float)xr[(size_t)t * DI];
      float du    = delta * xval;
      float y     = Dd * xval;
#pragma unroll
      for (int n = 0; n < NS; ++n) {
        h[n] = h[n] * __expf(delta * Areg[n]) + du * lds[(RK + n) * TCH + tt];
        y   += h[n] * lds[(RK + NS + n) * TCH + tt];
      }
      // inverse of windowed scan -> pixel; CrossMerge via atomic add
      int iw = t % wsz, r1 = t / wsz;
      int ih = r1 % wsz, g = r1 / wsz;
      int gw = g % gdiv, gh = g / gdiv;
      int pix = (gh * wsz + ih) * WWI + (gw * wsz + iw);
      atomicAdd(yrow + (size_t)pix * DI, y);
    }
  }
}

// ---------------- 6) channel LayerNorm + gate -> yg (f16, (b,l,d)) ----------------
__global__ __launch_bounds__(32) void k_ln_gate(const float* __restrict__ ybuf,
                                                const float* __restrict__ zbuf,
                                                const float* __restrict__ gamma,
                                                const float* __restrict__ beta,
                                                _Float16* __restrict__ yg) {
  int row = blockIdx.x;   // b*LL + l
  int lane = threadIdx.x;
  const float* yr = ybuf + (size_t)row * DI;
  float v[6]; float s = 0.0f, s2 = 0.0f;
#pragma unroll
  for (int j = 0; j < 6; ++j) { v[j] = yr[lane + 32 * j]; s += v[j]; s2 += v[j] * v[j]; }
#pragma unroll
  for (int off = 16; off > 0; off >>= 1) {
    s  += __shfl_xor(s,  off, 32);
    s2 += __shfl_xor(s2, off, 32);
  }
  float mu  = s * (1.0f / DI);
  float var = s2 * (1.0f / DI) - mu * mu;
  float inv = rsqrtf(var + 1e-5f);
  const float* zr = zbuf + (size_t)row * DI;
  _Float16* og = yg + (size_t)row * DI;
#pragma unroll
  for (int j = 0; j < 6; ++j) {
    int dd = lane + 32 * j;
    float tv = (v[j] - mu) * inv * gamma[dd] + beta[dd];
    og[dd] = (_Float16)(tv * zr[dd]);
  }
}

// ---------------- 7) out_proj: (BL x 192) @ (192 x 96) -> d_out ----------------
__global__ __launch_bounds__(32) void k_gemm_out(const _Float16* __restrict__ yg,
                                                 const _Float16* __restrict__ wouth,  // (96 x 192) N-major
                                                 float* __restrict__ out) {
  int lane = threadIdx.x;
  int m0 = blockIdx.x * 16;
  v8f acc[6] = {};
  wmma_rows<6>(yg, DI, BLR - 1, wouth, DI, m0, 0, DI, lane, acc);
  int ml = (lane >> 4) << 3, nl = lane & 15;
#pragma unroll
  for (int t = 0; t < 6; ++t) {
    int ng = 16 * t + nl;
#pragma unroll
    for (int v = 0; v < 8; ++v) {
      int mg = m0 + ml + v;
      out[(size_t)mg * DM + ng] = acc[t][v];
    }
  }
}

extern "C" void kernel_launch(void* const* d_in, const int* in_sizes, int n_in,
                              void* d_out, int out_size, void* d_ws, size_t ws_size,
                              hipStream_t stream) {
  (void)in_sizes; (void)n_in; (void)out_size; (void)ws_size;
  const float* x         = (const float*)d_in[0];
  const float* in_proj_w = (const float*)d_in[1];
  const float* conv_w    = (const float*)d_in[2];
  const float* conv_b    = (const float*)d_in[3];
  const float* x_proj_w  = (const float*)d_in[4];
  const float* dt_w      = (const float*)d_in[5];
  const float* dt_b      = (const float*)d_in[6];
  const float* A_logs    = (const float*)d_in[7];
  const float* Dsv       = (const float*)d_in[8];
  const float* ln_g      = (const float*)d_in[9];
  const float* ln_b      = (const float*)d_in[10];
  const float* out_w     = (const float*)d_in[11];
  float* out = (float*)d_out;

  char* ws = (char*)d_ws;
  size_t off = 0;
  auto take = [&](size_t bytes) { char* p = ws + off; off = (off + bytes + 255) & ~(size_t)255; return p; };
  _Float16* xh    = (_Float16*)take((size_t)BLR * DM * 2);          // x as f16 (BL x 96)
  _Float16* winh  = (_Float16*)take((size_t)2 * DI * DM * 2);       // in_proj_w f16 (384 x 96) N-major
  _Float16* xpwh  = (_Float16*)take((size_t)KG * CDBL * DI * 2);    // x_proj_weight f16 (K-major A)
  _Float16* wouth = (_Float16*)take((size_t)DM * DI * 2);           // out_proj_w f16 (96 x 192) N-major
  float*    xc_pre= (float*)take((size_t)BS * DI * LL * 4);         // pre-conv (b,d,h,w)
  float*    zbuf  = (float*)take((size_t)BLR * DI * 4);             // silu gate (b,l,d)
  float*    xc    = (float*)take((size_t)BS * DI * LL * 4);         // post-conv (b,d,h,w)
  _Float16* xsh   = (_Float16*)take((size_t)BS * KG * LL * DI * 2); // xs f16 (b,k,l,d)
  float*    xdbl  = (float*)take((size_t)BS * KG * CDBL * LL * 4);  // dt/B/C rows (b,k,c,l)
  float*    ybuf  = (float*)take((size_t)BLR * DI * 4);             // merged y (b,l,d)
  _Float16* yg    = (_Float16*)take((size_t)BLR * DI * 2);          // LN+gated f16

  const int TB = 256;
  k_f32_to_f16<<<dim3((BLR * DM + TB - 1) / TB), dim3(TB), 0, stream>>>(x, xh, BLR * DM);
  k_f32_to_f16<<<dim3((2 * DI * DM + TB - 1) / TB), dim3(TB), 0, stream>>>(in_proj_w, winh, 2 * DI * DM);
  k_f32_to_f16<<<dim3((KG * CDBL * DI + TB - 1) / TB), dim3(TB), 0, stream>>>(x_proj_w, xpwh, KG * CDBL * DI);
  k_f32_to_f16<<<dim3((DM * DI + TB - 1) / TB), dim3(TB), 0, stream>>>(out_w, wouth, DM * DI);

  k_gemm_inproj<<<dim3(BLR / 16, (2 * DI) / 64), dim3(32), 0, stream>>>(xh, winh, xc_pre, zbuf);
  k_conv_dw    <<<dim3((BS * DI * LL + TB - 1) / TB), dim3(TB), 0, stream>>>(xc_pre, conv_w, conv_b, xc);
  k_cross_scan <<<dim3((BS * DI * LL + TB - 1) / TB), dim3(TB), 0, stream>>>(xc, xsh);
  k_gemm_xproj <<<dim3(LL / 64, 3, BS * KG), dim3(32), 0, stream>>>(xpwh, xsh, xdbl);
  k_zero       <<<dim3((BLR * DI + TB - 1) / TB), dim3(TB), 0, stream>>>(ybuf, BLR * DI);
  k_scan       <<<dim3(BS * KG * 6), dim3(32), 0, stream>>>(xdbl, xsh, dt_w, dt_b, A_logs, Dsv, ybuf);
  k_ln_gate    <<<dim3(BLR), dim3(32), 0, stream>>>(ybuf, zbuf, ln_g, ln_b, yg);
  k_gemm_out   <<<dim3(BLR / 16, 1), dim3(32), 0, stream>>>(yg, wouth, out);
}
